// LinkPredictionModel_GIN_37709812859563
// MI455X (gfx1250) — matrix operand order repaired and checked by reading, not compiled, not run on hardware
//
#include <hip/hip_runtime.h>
#include <hip/hip_bf16.h>
#include <math.h>

// ---------------------------------------------------------------------------
// GIN 2-layer forward for MI455X (gfx1250, wave32, WMMA).
// Memory-dominated workload; GEMM done with v_wmma_f32_16x16x32_bf16 using a
// bf16 hi/lo split of the f32 operands for near-fp32 accuracy.
// ---------------------------------------------------------------------------

typedef __attribute__((ext_vector_type(8)))  float  v8f;
typedef __attribute__((ext_vector_type(16))) __bf16 v16bf;
typedef __attribute__((ext_vector_type(8)))  __bf16 v8bf;

union V16 { v16bf v; v8bf h[2]; };

#define FEAT 128
#define BN_EPS 1e-5f

// ---------------------------------------------------------------------------
// 1) float4 copy: agg = x  (folds GIN's "+ x_i" self term into the scatter buf)
// ---------------------------------------------------------------------------
__global__ __launch_bounds__(256) void copy_f4_kernel(
    const float4* __restrict__ in, float4* __restrict__ out, int n4) {
  int i = blockIdx.x * 256 + threadIdx.x;
  if (i < n4) out[i] = in[i];
}

// ---------------------------------------------------------------------------
// 2) scatter-add: agg[dst[e]][:] += X[src[e]][:]
//    One wave (32 lanes) per edge; lane handles features f, f+32, f+64, f+96.
//    Coalesced 128B gathers + coalesced global_atomic_add_f32 (L2-resident).
// ---------------------------------------------------------------------------
__global__ __launch_bounds__(256) void scatter_add_kernel(
    const float* __restrict__ X, const int* __restrict__ src,
    const int* __restrict__ dst, float* __restrict__ agg, int E) {
  int e = blockIdx.x * 8 + (threadIdx.x >> 5);
  if (e >= E) return;
  int lane = threadIdx.x & 31;
  int s = src[e], d = dst[e];
  const float* xs = X + (size_t)s * FEAT;
  float*       ad = agg + (size_t)d * FEAT;
#pragma unroll
  for (int i = 0; i < 4; ++i) {
    int f = lane + 32 * i;
    atomicAdd(&ad[f], xs[f]);
  }
}

// ---------------------------------------------------------------------------
// 3) zero the BN stat accumulators (colsum[128], colsumsq[128])
// ---------------------------------------------------------------------------
__global__ void zero_stats_kernel(float* __restrict__ stats) {
  stats[threadIdx.x] = 0.0f;  // 256 threads -> 256 floats
}

// ---------------------------------------------------------------------------
// 4) Fused GEMM + bias + ReLU + BN-stat accumulation.
//    H = relu(A @ W + b);  atomically accumulate per-column sum / sum^2.
//    Block: 256 threads = 8 waves; block covers rows [16*bid, 16*bid+16),
//    wave w covers cols [16w, 16w+16). K = 128 in 4 steps of 32 (bf16 WMMA).
//    fp32 emulated as Ahi*Bhi + Ahi*Blo + Alo*Bhi (fp32 accumulate).
// ---------------------------------------------------------------------------
__global__ __launch_bounds__(256) void gemm_bnstats_kernel(
    const float* __restrict__ A,     // [N,128] = x + sum_neighbors
    const float* __restrict__ W,     // [128,128] row-major (k, n)
    const float* __restrict__ bias,  // [128]
    float* __restrict__ H,           // [N,128] out, post-ReLU
    float* __restrict__ colsum,      // [128]
    float* __restrict__ colsumsq) {  // [128]
  __shared__ __bf16 whi[FEAT][FEAT];  // transposed: [n][k]
  __shared__ __bf16 wlo[FEAT][FEAT];
  __shared__ __bf16 ahi[16][FEAT];    // [m][k]
  __shared__ __bf16 alo[16][FEAT];

  const int t    = threadIdx.x;
  const int row0 = blockIdx.x * 16;

  // Stage W -> LDS, transposed, split into bf16 hi/lo. Coalesced global reads.
#pragma unroll 4
  for (int i = 0; i < 64; ++i) {
    int idx = t + 256 * i;          // 0..16383
    int k = idx >> 7, n = idx & 127;
    float w   = W[idx];
    __bf16 hi = (__bf16)w;
    whi[n][k] = hi;
    wlo[n][k] = (__bf16)(w - (float)hi);
  }
  // Stage the 16x128 A tile -> LDS (hi/lo). Coalesced reads & LDS writes.
#pragma unroll
  for (int i = 0; i < 8; ++i) {
    int idx = t + 256 * i;          // 0..2047
    int m = idx >> 7, k = idx & 127;
    float a   = A[(size_t)(row0 + m) * FEAT + k];
    __bf16 hi = (__bf16)a;
    ahi[m][k] = hi;
    alo[m][k] = (__bf16)(a - (float)hi);
  }
  __syncthreads();

  const int wave = t >> 5;   // 0..7 -> column tile
  const int lane = t & 31;
  const int nn   = lane & 15;
  const int half = lane >> 4;
  const int n0   = wave * 16;

  v8f c = {};
#pragma unroll
  for (int kb = 0; kb < 4; ++kb) {
    // A fragment (16x32 bf16): lane row = nn; K runs {h*8..+7} and {16+h*8..+7}
    const int ka = kb * 32 + half * 8;
    V16 a_hi, a_lo, b_hi, b_lo;
    a_hi.h[0] = *(const v8bf*)&ahi[nn][ka];
    a_hi.h[1] = *(const v8bf*)&ahi[nn][ka + 16];
    a_lo.h[0] = *(const v8bf*)&alo[nn][ka];
    a_lo.h[1] = *(const v8bf*)&alo[nn][ka + 16];
    // B fragment (32x16 bf16): lane col = n0+nn; K runs {h*8..+7} and {16+h*8..+7}
    const int kbm = kb * 32 + half * 8;
    b_hi.h[0] = *(const v8bf*)&whi[n0 + nn][kbm];
    b_hi.h[1] = *(const v8bf*)&whi[n0 + nn][kbm + 16];
    b_lo.h[0] = *(const v8bf*)&wlo[n0 + nn][kbm];
    b_lo.h[1] = *(const v8bf*)&wlo[n0 + nn][kbm + 16];

    c = __builtin_amdgcn_wmma_f32_16x16x32_bf16(false, a_hi.v, false, b_hi.v,
                                                (short)0, c, false, false);
    c = __builtin_amdgcn_wmma_f32_16x16x32_bf16(false, a_hi.v, false, b_lo.v,
                                                (short)0, c, false, false);
    c = __builtin_amdgcn_wmma_f32_16x16x32_bf16(false, a_lo.v, false, b_hi.v,
                                                (short)0, c, false, false);
  }

  // Epilogue: bias + ReLU, store, fused BN statistics.
  // C/D layout: VGPR r, lane l -> row = r + 8*(l>=16), col = l%16.
  const int   col  = n0 + nn;
  const float bcol = bias[col];
  float s = 0.0f, s2 = 0.0f;
#pragma unroll
  for (int r = 0; r < 8; ++r) {
    float v = c[r] + bcol;
    v = v > 0.0f ? v : 0.0f;
    int row = row0 + r + half * 8;
    H[(size_t)row * FEAT + col] = v;
    s  += v;
    s2 += v * v;
  }
  atomicAdd(&colsum[col], s);
  atomicAdd(&colsumsq[col], s2);
}

// ---------------------------------------------------------------------------
// 5) BN finalize: scale = g*rsqrt(var+eps), shift = beta - mean*scale
// ---------------------------------------------------------------------------
__global__ void bn_finalize_kernel(const float* __restrict__ colsum,
                                   const float* __restrict__ colsumsq,
                                   const float* __restrict__ gamma,
                                   const float* __restrict__ beta,
                                   float* __restrict__ scale,
                                   float* __restrict__ shift, float invN) {
  int c = threadIdx.x;  // 128 threads
  float mean = colsum[c] * invN;
  float var  = colsumsq[c] * invN - mean * mean;
  float sc   = gamma[c] * rsqrtf(var + BN_EPS);
  scale[c]   = sc;
  shift[c]   = beta[c] - mean * sc;
}

// ---------------------------------------------------------------------------
// 6) BN apply: out = h*scale[col] + shift[col], float4-vectorized
// ---------------------------------------------------------------------------
__global__ __launch_bounds__(256) void bn_apply_kernel(
    const float4* __restrict__ h, const float* __restrict__ scale,
    const float* __restrict__ shift, float4* __restrict__ out, int n4) {
  int i = blockIdx.x * 256 + threadIdx.x;
  if (i >= n4) return;
  float4 v = h[i];
  int c = (i & 31) * 4;  // 128 cols / 4 per float4
  float4 o;
  o.x = v.x * scale[c + 0] + shift[c + 0];
  o.y = v.y * scale[c + 1] + shift[c + 1];
  o.z = v.z * scale[c + 2] + shift[c + 2];
  o.w = v.w * scale[c + 3] + shift[c + 3];
  out[i] = o;
}

// ---------------------------------------------------------------------------
// Host launcher
// ---------------------------------------------------------------------------
extern "C" void kernel_launch(void* const* d_in, const int* in_sizes, int n_in,
                              void* d_out, int out_size, void* d_ws,
                              size_t ws_size, hipStream_t stream) {
  const float* x    = (const float*)d_in[0];
  const int*   ei   = (const int*)d_in[1];  // [2, E] int32
  const float* W1   = (const float*)d_in[2];
  const float* b1   = (const float*)d_in[3];
  const float* W2   = (const float*)d_in[4];
  const float* b2   = (const float*)d_in[5];
  const float* g1   = (const float*)d_in[6];
  const float* be1  = (const float*)d_in[7];
  const float* g2   = (const float*)d_in[8];
  const float* be2  = (const float*)d_in[9];
  float*       out  = (float*)d_out;

  const int N = in_sizes[0] / FEAT;  // 50000
  const int E = in_sizes[1] / 2;     // 800000
  const int* src = ei;
  const int* dst = ei + E;

  const size_t NF = (size_t)N * FEAT;
  float* agg   = (float*)d_ws;       // [N,128]
  float* h     = agg + NF;           // [N,128]
  float* hn    = h + NF;             // [N,128]
  float* stats = hn + NF;            // colsum[128], colsumsq[128]
  float* scale = stats + 256;        // [128]
  float* shift = scale + 128;        // [128]

  const int n4        = (int)(NF / 4);
  const int copyBlks  = (n4 + 255) / 256;
  const int scatBlks  = (E + 7) / 8;
  const int gemmBlks  = N / 16;      // 50000/16 = 3125 exact
  const float invN    = 1.0f / (float)N;

  // ---- Layer 1 ----
  copy_f4_kernel<<<copyBlks, 256, 0, stream>>>((const float4*)x, (float4*)agg, n4);
  scatter_add_kernel<<<scatBlks, 256, 0, stream>>>(x, src, dst, agg, E);
  zero_stats_kernel<<<1, 256, 0, stream>>>(stats);
  gemm_bnstats_kernel<<<gemmBlks, 256, 0, stream>>>(agg, W1, b1, h, stats,
                                                    stats + 128);
  bn_finalize_kernel<<<1, 128, 0, stream>>>(stats, stats + 128, g1, be1, scale,
                                            shift, invN);
  bn_apply_kernel<<<copyBlks, 256, 0, stream>>>((const float4*)h, scale, shift,
                                                (float4*)hn, n4);

  // ---- Layer 2 ----
  copy_f4_kernel<<<copyBlks, 256, 0, stream>>>((const float4*)hn, (float4*)agg, n4);
  scatter_add_kernel<<<scatBlks, 256, 0, stream>>>(hn, src, dst, agg, E);
  zero_stats_kernel<<<1, 256, 0, stream>>>(stats);
  gemm_bnstats_kernel<<<gemmBlks, 256, 0, stream>>>(agg, W2, b2, h, stats,
                                                    stats + 128);
  bn_finalize_kernel<<<1, 128, 0, stream>>>(stats, stats + 128, g2, be2, scale,
                                            shift, invN);
  bn_apply_kernel<<<copyBlks, 256, 0, stream>>>((const float4*)h, scale, shift,
                                                (float4*)out, n4);
}
